// CharLSTMEmbedding_50079318671863
// MI455X (gfx1250) — matrix operand order repaired and checked by reading, not compiled, not run on hardware
//
#include <hip/hip_runtime.h>
#include <hip/hip_bf16.h>

typedef __attribute__((ext_vector_type(16))) _Float16 v16h;
typedef __attribute__((ext_vector_type(8)))  _Float16 v8h;
typedef __attribute__((ext_vector_type(8)))  float    v8f;

#define BT_   4096      // B*T rows
#define LSEQ_ 16
#define EDIM_ 256
#define HDIM_ 512
#define KDIM_ 768       // E + H  (concatenated GEMM reduction dim)
#define GDIM_ 2048      // 4*H    (i,f,g,o)
#define VOCAB_ 256

#define KCH_PHASE 12    // 32-wide K-chunks per LDS phase (2 phases x 12 x 32 = 768)

// ---------- fast activations (safe at large |x|) ----------
__device__ __forceinline__ float fsigmoid(float x) {
    return 1.0f / (1.0f + __expf(-x));
}
__device__ __forceinline__ float ftanh(float x) {
    float e = __expf(-2.0f * fabsf(x));        // in (0,1]
    float r = (1.0f - e) / (1.0f + e);
    return copysignf(r, x);
}

__device__ __forceinline__ v16h pack16(v8h lo, v8h hi) {
    v16h r;
#pragma unroll
    for (int i = 0; i < 8; ++i) { r[i] = lo[i]; r[8 + i] = hi[i]; }
    return r;
}

// ---------- one-time conversion kernels (run every call; deterministic) ----------
__global__ void k_convert_w(const float* __restrict__ Wih,
                            const float* __restrict__ Whh,
                            _Float16* __restrict__ Wcomb) {
    int idx = blockIdx.x * blockDim.x + threadIdx.x;
    if (idx >= GDIM_ * KDIM_) return;
    int row = idx / KDIM_;
    int col = idx - row * KDIM_;
    float v = (col < EDIM_) ? Wih[row * EDIM_ + col]
                            : Whh[row * HDIM_ + (col - EDIM_)];
    Wcomb[idx] = (_Float16)v;
}

__global__ void k_convert_emb_bias(const float* __restrict__ emb,
                                   const float* __restrict__ bih,
                                   const float* __restrict__ bhh,
                                   _Float16* __restrict__ embH,
                                   float* __restrict__ biasC) {
    int idx = blockIdx.x * blockDim.x + threadIdx.x;
    if (idx < VOCAB_ * EDIM_) embH[idx] = (_Float16)emb[idx];
    if (idx < GDIM_)          biasC[idx] = bih[idx] + bhh[idx];
}

__global__ void k_init_hc(float* __restrict__ h, float* __restrict__ c) {
    int idx = blockIdx.x * blockDim.x + threadIdx.x;
    if (idx >= BT_ * HDIM_) return;
    h[idx] = 0.0f;
    c[idx] = 0.0f;
}

// ---------- per-step: build A = [ emb[ids_t] | f16(h_prev) ]  (row-major [4096][768] f16) ----------
__global__ void k_prepare(const int* __restrict__ seq,      // [BT][L]
                          const float* __restrict__ hF32,   // [BT][H]
                          const _Float16* __restrict__ embH,// [VOCAB][E]
                          _Float16* __restrict__ Abuf,      // [BT][K]
                          int t) {
    int m   = blockIdx.x;           // row 0..4095
    int tid = threadIdx.x;          // 0..255
    int id  = seq[m * LSEQ_ + t];
    _Float16* arow = Abuf + (size_t)m * KDIM_;
    const float* hrow = hF32 + (size_t)m * HDIM_;
    arow[tid]               = embH[id * EDIM_ + tid];     // x part (E=256)
    arow[EDIM_ + tid]       = (_Float16)hrow[tid];        // h part (H=512)
    arow[EDIM_ + 256 + tid] = (_Float16)hrow[256 + tid];
}

// ---------- fused LSTM step: WMMA GEMM, LDS-staged weights, software-pipelined fragments ----------
// grid = (16, 32): blockIdx.x -> 256 M-rows (8 waves x 2 tiles x 16), blockIdx.y -> 16 H-cols
// LDS holds the block's B slice (4 gates x 16 W-rows) for half of K at a time, pre-swizzled
// into WMMA fragment order: per (gate, kchunk): [lo: 32 lanes x 16B][hi: 32 lanes x 16B].
__global__ __launch_bounds__(256) void k_lstm_step(
    const _Float16* __restrict__ Abuf,   // [BT][K] f16
    const _Float16* __restrict__ Wcomb,  // [4H][K] f16
    const float*    __restrict__ biasC,  // [4H]
    const int*      __restrict__ lens,   // [BT]
    float* __restrict__ hF32,            // [BT][H] in/out
    float* __restrict__ cF32,            // [BT][H] in/out
    int t) {
    __shared__ _Float16 Bs[4 * KCH_PHASE * 512];   // 48 KB

    const int tid  = threadIdx.x;
    const int lane = tid & 31;
    const int wave = tid >> 5;
    const int m0   = blockIdx.x * 256 + wave * 32;   // two 16-row tiles: m0, m0+16
    const int n0   = blockIdx.y * 16;
    const int half = lane >> 4;          // 0: lanes 0-15, 1: lanes 16-31
    const int l15  = lane & 15;

    v8f acc[2][4] = {};                  // [m-tile][gate]

    // A fragments: lane holds row M; half 0 -> K chunks {0..7,16..23}, half 1 -> {8..15,24..31}
    const _Float16* Arow0 = Abuf + (size_t)(m0 + l15) * KDIM_ + half * 8;
    const _Float16* Arow1 = Arow0 + (size_t)16 * KDIM_;
    // per-lane B fragment addresses inside a (gate,kchunk) slot
    const _Float16* BsLo = Bs + lane * 8;
    const _Float16* BsHi = Bs + 256 + lane * 8;

#pragma unroll
    for (int ph = 0; ph < 2; ++ph) {
        // ---- cooperative preload of B slice for this K-phase into LDS ----
        for (int fr = tid; fr < 4 * KCH_PHASE * 32; fr += 256) {
            const int g   = fr / (KCH_PHASE * 32);
            const int rem = fr - g * (KCH_PHASE * 32);
            const int kc  = rem >> 5;
            const int ls  = rem & 31;
            const int row = g * HDIM_ + n0 + (ls & 15);
            const int kg  = ph * (KCH_PHASE * 32) + kc * 32 + (ls >> 4) * 8;
            const v8h* src = (const v8h*)(Wcomb + (size_t)row * KDIM_ + kg);
            _Float16* fb = Bs + (g * KCH_PHASE + kc) * 512;   // fragment base (f16 units)
            *(v8h*)(fb + ls * 8)       = src[0];              // lo 16B, lane-stride 16B
            *(v8h*)(fb + 256 + ls * 8) = src[2];              // hi 16B
        }
        __syncthreads();

        const int kb = ph * (KCH_PHASE * 32);

        // ---- prime the software pipeline with chunk 0's fragments ----
        v16h a0c = pack16(*(const v8h*)(Arow0 + kb), *(const v8h*)(Arow0 + kb + 16));
        v16h a1c = pack16(*(const v8h*)(Arow1 + kb), *(const v8h*)(Arow1 + kb + 16));
        v16h bc[4];
#pragma unroll
        for (int g = 0; g < 4; ++g) {
            const int fo = (g * KCH_PHASE) * 512;
            bc[g] = pack16(*(const v8h*)(BsLo + fo), *(const v8h*)(BsHi + fo));
        }

        // ---- pipelined compute: issue chunk kc+1 loads before chunk kc's 8 WMMAs ----
#pragma unroll
        for (int kc = 0; kc < KCH_PHASE; ++kc) {
            v16h a0n, a1n, bn[4];
            if (kc + 1 < KCH_PHASE) {
                const int kg = kb + (kc + 1) * 32;
                a0n = pack16(*(const v8h*)(Arow0 + kg), *(const v8h*)(Arow0 + kg + 16));
                a1n = pack16(*(const v8h*)(Arow1 + kg), *(const v8h*)(Arow1 + kg + 16));
#pragma unroll
                for (int g = 0; g < 4; ++g) {
                    const int fo = (g * KCH_PHASE + kc + 1) * 512;
                    bn[g] = pack16(*(const v8h*)(BsLo + fo), *(const v8h*)(BsHi + fo));
                }
            }
#pragma unroll
            for (int g = 0; g < 4; ++g) {
                acc[0][g] = __builtin_amdgcn_wmma_f32_16x16x32_f16(
                    false, a0c, false, bc[g], (short)0, acc[0][g], false, false);
                acc[1][g] = __builtin_amdgcn_wmma_f32_16x16x32_f16(
                    false, a1c, false, bc[g], (short)0, acc[1][g], false, false);
            }
            if (kc + 1 < KCH_PHASE) {
                a0c = a0n; a1c = a1n;
#pragma unroll
                for (int g = 0; g < 4; ++g) bc[g] = bn[g];
            }
        }
        __syncthreads();   // protect LDS before next phase's preload
    }

    // ---- epilogue: all four gates for both 16x16 H-tiles are in registers ----
    // C/D layout: VGPR r, lanes 0-15 -> M=r ; lanes 16-31 -> M=r+8 ; N = lane&15.
    const int col = n0 + l15;
    const float bI = biasC[col];
    const float bF = biasC[HDIM_ + col];
    const float bG = biasC[2 * HDIM_ + col];
    const float bO = biasC[3 * HDIM_ + col];
#pragma unroll
    for (int mt = 0; mt < 2; ++mt) {
#pragma unroll
        for (int r = 0; r < 8; ++r) {
            const int m = m0 + mt * 16 + r + half * 8;
            const float iV = fsigmoid(acc[mt][0][r] + bI);
            const float fV = fsigmoid(acc[mt][1][r] + bF);
            const float gV = ftanh  (acc[mt][2][r] + bG);
            const float oV = fsigmoid(acc[mt][3][r] + bO);
            const size_t off = (size_t)m * HDIM_ + col;
            const float cOld = cF32[off];
            const float cNew = fV * cOld + iV * gV;
            const float hNew = oV * ftanh(cNew);
            const bool active = lens[m] > t;             // length-mask blend
            const float hOld = hF32[off];
            hF32[off] = active ? hNew : hOld;
            cF32[off] = active ? cNew : cOld;
        }
    }
}

// ---------- host launcher ----------
extern "C" void kernel_launch(void* const* d_in, const int* in_sizes, int n_in,
                              void* d_out, int out_size, void* d_ws, size_t ws_size,
                              hipStream_t stream) {
    (void)in_sizes; (void)n_in; (void)out_size; (void)ws_size;
    const int*   seq  = (const int*)  d_in[0];  // [B,T,L] int32
    const int*   lens = (const int*)  d_in[1];  // [B,T]   int32
    const float* emb  = (const float*)d_in[2];  // [256,256]
    const float* Wih  = (const float*)d_in[3];  // [2048,256]
    const float* Whh  = (const float*)d_in[4];  // [2048,512]
    const float* bih  = (const float*)d_in[5];  // [2048]
    const float* bhh  = (const float*)d_in[6];  // [2048]

    char* ws = (char*)d_ws;
    constexpr size_t OFF_EMBH  = 0;                                   // 256*256*2   = 128 KB
    constexpr size_t OFF_WCOMB = OFF_EMBH  + (size_t)VOCAB_*EDIM_*2;  // 2048*768*2  = 3 MB
    constexpr size_t OFF_BIAS  = OFF_WCOMB + (size_t)GDIM_*KDIM_*2;   // 2048*4      = 8 KB
    constexpr size_t OFF_ABUF  = OFF_BIAS  + (size_t)GDIM_*4;         // 4096*768*2  = 6 MB
    constexpr size_t OFF_H     = OFF_ABUF  + (size_t)BT_*KDIM_*2;     // 4096*512*4  = 8 MB
    constexpr size_t OFF_C     = OFF_H     + (size_t)BT_*HDIM_*4;     // 8 MB
    _Float16* embH  = (_Float16*)(ws + OFF_EMBH);
    _Float16* Wcomb = (_Float16*)(ws + OFF_WCOMB);
    float*    biasC = (float*)   (ws + OFF_BIAS);
    _Float16* Abuf  = (_Float16*)(ws + OFF_ABUF);
    float*    hF32  = (float*)   (ws + OFF_H);
    float*    cF32  = (float*)   (ws + OFF_C);

    k_convert_w<<<(GDIM_ * KDIM_ + 255) / 256, 256, 0, stream>>>(Wih, Whh, Wcomb);
    k_convert_emb_bias<<<(VOCAB_ * EDIM_ + 255) / 256, 256, 0, stream>>>(emb, bih, bhh, embH, biasC);
    k_init_hc<<<(BT_ * HDIM_ + 255) / 256, 256, 0, stream>>>(hF32, cF32);

    dim3 gstep(16, 32);   // 16 M-blocks (256 rows) x 32 N-blocks (16 H-cols)
    for (int t = 0; t < LSEQ_; ++t) {
        k_prepare<<<BT_, 256, 0, stream>>>(seq, hF32, embH, Abuf, t);
        k_lstm_step<<<gstep, 256, 0, stream>>>(Abuf, Wcomb, biasC, lens, hF32, cF32, t);
    }

    // Final hidden state is exactly the output [B,T,H] = [4096,512] f32.
    hipMemcpyAsync(d_out, hF32, (size_t)BT_ * HDIM_ * sizeof(float),
                   hipMemcpyDeviceToDevice, stream);
}
